// Vocoder_55611236549452
// MI455X (gfx1250) — matrix operand order repaired and checked by reading, not compiled, not run on hardware
//
#include <hip/hip_runtime.h>
#include <hip/hip_bf16.h>
#include <math.h>
#include <stdint.h>

typedef __attribute__((ext_vector_type(16))) _Float16 v16h;
typedef __attribute__((ext_vector_type(8)))  _Float16 v8h;
typedef __attribute__((ext_vector_type(8)))  float    v8f;

#define MEL   80
#define COND  128
#define EMBD  256
#define RNN   896
#define FCD   512
#define QD    512
#define FPS   40
#define HOP   200
#define PADF  16
#define BB    32
#define TT    1600
#define G3    (3*RNN)        /* 2688 */
#define XDIM  (EMBD+2*COND)  /* 512  */

__device__ __forceinline__ float sigmoidf_(float x) { return 1.0f / (1.0f + __expf(-x)); }

// ---------------------------------------------------------------------------
// f32 -> f16 weight conversion
// ---------------------------------------------------------------------------
__global__ void f32_to_f16_kernel(const float* __restrict__ src, _Float16* __restrict__ dst, int n) {
    for (int i = blockIdx.x * blockDim.x + threadIdx.x; i < n; i += gridDim.x * blockDim.x)
        dst[i] = (_Float16)src[i];
}

// ---------------------------------------------------------------------------
// Conditioning GRU (tiny: B=32 blocks x 128 threads, one hidden unit/thread,
// h kept in LDS, 40 serial steps). reverse=1 processes frames backward.
// ---------------------------------------------------------------------------
__global__ void cond_gru_kernel(const float* __restrict__ x, int in_dim,
                                const float* __restrict__ w_ih, const float* __restrict__ w_hh,
                                const float* __restrict__ b_ih, const float* __restrict__ b_hh,
                                float* __restrict__ out, int col_off, int reverse) {
    __shared__ float x_s[2 * COND];
    __shared__ float h_s[COND];
    const int b = blockIdx.x;
    const int j = threadIdx.x;           // 0..127
    h_s[j] = 0.f;
    __syncthreads();
    for (int s = 0; s < FPS; ++s) {
        const int t = reverse ? (FPS - 1 - s) : s;
        for (int i = j; i < in_dim; i += COND)
            x_s[i] = x[((size_t)b * FPS + t) * in_dim + i];
        __syncthreads();
        float xr = b_ih[j], xz = b_ih[COND + j], xn = b_ih[2 * COND + j];
        const float* wr = w_ih + (size_t)j * in_dim;
        const float* wz = w_ih + (size_t)(COND + j) * in_dim;
        const float* wn = w_ih + (size_t)(2 * COND + j) * in_dim;
        for (int i = 0; i < in_dim; ++i) {
            const float xv = x_s[i];
            xr += wr[i] * xv; xz += wz[i] * xv; xn += wn[i] * xv;
        }
        float hr = b_hh[j], hz = b_hh[COND + j], hn = b_hh[2 * COND + j];
        const float* ur = w_hh + (size_t)j * COND;
        const float* uz = w_hh + (size_t)(COND + j) * COND;
        const float* un = w_hh + (size_t)(2 * COND + j) * COND;
        for (int i = 0; i < COND; ++i) {
            const float hv = h_s[i];
            hr += ur[i] * hv; hz += uz[i] * hv; hn += un[i] * hv;
        }
        const float r = sigmoidf_(xr + hr);
        const float z = sigmoidf_(xz + hz);
        const float n = tanhf(xn + r * hn);
        const float hnew = (1.f - z) * n + z * h_s[j];   // read old h before barrier
        __syncthreads();
        h_s[j] = hnew;
        out[((size_t)b * FPS + t) * (2 * COND) + col_off + j] = hnew;
        __syncthreads();
    }
}

// ---------------------------------------------------------------------------
// Build x = concat(emb[wav], upsampled cond) as f16, rows r = b*T + t, K=512
// ---------------------------------------------------------------------------
__global__ void build_x_kernel(const int* __restrict__ wavs, const float* __restrict__ emb,
                               const float* __restrict__ cond, _Float16* __restrict__ x) {
    const int r = blockIdx.x;            // b*T + t
    const int b = r / TT, t = r % TT;
    const int j = threadIdx.x;           // 0..255
    const int w = wavs[r];
    const int f = PADF + t / HOP;
    x[(size_t)r * XDIM + j]        = (_Float16)emb[(size_t)w * EMBD + j];
    x[(size_t)r * XDIM + EMBD + j] = (_Float16)cond[((size_t)b * FPS + f) * (2 * COND) + j];
}

// ---------------------------------------------------------------------------
// WMMA f16 GEMM:  C[M,N] = A[M,K] @ B[N,K]^T (+bias) (+relu)
// 8 waves/block share one M-tile (requires tilesN % 8 == 0, true for all
// three GEMMs: 168, 32, 32). The block's 16xK A strip is staged into LDS once
// with GLOBAL_LOAD_ASYNC_TO_LDS_B128 (ASYNCcnt), then every wave reads A
// fragments from LDS and streams its own B tile from global/L2.
// ---------------------------------------------------------------------------
template <typename OutT, bool RELU>
__global__ void wmma_gemm_kernel(const _Float16* __restrict__ A, const _Float16* __restrict__ Bw,
                                 const float* __restrict__ bias, OutT* __restrict__ C,
                                 int M, int N, int K) {
    extern __shared__ _Float16 a_s[];    // 16 x K
    const int lane   = threadIdx.x & 31;
    const int wave   = threadIdx.x >> 5;
    const int tilesN = N >> 4;
    const int tile0  = blockIdx.x * 8;               // first tile of this block
    const int tm     = tile0 / tilesN;               // uniform across block
    const int tn     = (tile0 % tilesN) + wave;      // per-wave N tile
    (void)M;

    // ---- async-stage A strip (16 x K f16) into LDS, 16B chunks per lane ----
    const int chunksPerRow = K >> 3;                 // 8 halves = 16 bytes
    const int totalChunks  = 16 * chunksPerRow;      // multiple of 256 for K=512/896
    for (int c = threadIdx.x; c < totalChunks; c += blockDim.x) {
        const int m = c / chunksPerRow;
        const int o = (c % chunksPerRow) << 3;
        const _Float16* gp = A + (size_t)(tm * 16 + m) * K + o;
        const unsigned  lp = (unsigned)(uintptr_t)(a_s + m * K + o); // AS3 offset = addr[31:0]
        asm volatile("global_load_async_to_lds_b128 %0, %1, off"
                     :: "v"(lp), "v"(gp) : "memory");
    }
    asm volatile("s_wait_asynccnt 0" ::: "memory");
    __syncthreads();

    // ---- compute: A fragments from LDS, B fragments from global ----
    const int l15 = lane & 15, grp = lane >> 4;
    const _Float16* arow = a_s + (size_t)l15 * K;
    const _Float16* brow = Bw + (size_t)(tn * 16 + l15) * K;
    v8f acc = {};
    for (int kk = 0; kk < K; kk += 32) {
        if (kk + 32 < K) __builtin_prefetch(brow + kk + 32 + 16 * grp, 0, 1);
        const v8h  alo = *(const v8h*)(arow + kk + 8 * grp);
        const v8h  ahi = *(const v8h*)(arow + kk + 16 + 8 * grp);
        const v16h a   = __builtin_shufflevector(alo, ahi, 0,1,2,3,4,5,6,7,8,9,10,11,12,13,14,15);
        const v8h  blo = *(const v8h*)(brow + kk + 8 * grp);
        const v8h  bhi = *(const v8h*)(brow + kk + 16 + 8 * grp);
        const v16h bm  = __builtin_shufflevector(blo, bhi, 0,1,2,3,4,5,6,7,8,9,10,11,12,13,14,15);
        acc = __builtin_amdgcn_wmma_f32_16x16x32_f16(false, a, false, bm, (short)0, acc,
                                                     false, false);
    }
    const int col = tn * 16 + l15;
    const float bv = bias[col];
#pragma unroll
    for (int v = 0; v < 8; ++v) {
        float val = acc[v] + bv;
        if (RELU) val = val > 0.f ? val : 0.f;
        C[(size_t)(tm * 16 + 8 * grp + v) * N + col] = (OutT)val;
    }
}

// ---------------------------------------------------------------------------
// Persistent recurrent GRU: 2 blocks (one per batch half, M=16), 1024 threads
// = 32 waves. h (16x896 f16) and gh (16x2688 f16) resident in LDS (114 KB of
// the 320 KB/WGP). Per step: 168 N-tiles x 28 K-step WMMAs, then fused gate
// update. gx was precomputed by the big GEMM; w_hh f16 streams from L2.
// ---------------------------------------------------------------------------
__global__ void gru_recurrent_kernel(const _Float16* __restrict__ gx,   // (B*T, 2688) f16, +b_ih
                                     const _Float16* __restrict__ whh,  // (2688, 896) f16
                                     const float* __restrict__ b_hh,    // (2688) f32
                                     _Float16* __restrict__ hs) {       // (B*T, 896) f16 out
    extern __shared__ _Float16 smem[];
    _Float16* h_s  = smem;            // 16*RNN
    _Float16* gh_s = smem + 16 * RNN; // 16*G3
    const int bm   = blockIdx.x * 16;
    const int tid  = threadIdx.x;
    const int lane = tid & 31, wave = tid >> 5;
    const int l15  = lane & 15, grp = lane >> 4;
    const int nwaves = blockDim.x >> 5;

    for (int i = tid; i < 16 * RNN; i += blockDim.x) h_s[i] = (_Float16)0.f;
    __syncthreads();

    for (int t = 0; t < TT; ++t) {
        // gh = h @ whhT : A fragments come from LDS (ds_load_b128)
        const _Float16* arow = h_s + (size_t)l15 * RNN;
        for (int tn = wave; tn < G3 / 16; tn += nwaves) {   // wave-uniform loop
            const _Float16* brow = whh + (size_t)(tn * 16 + l15) * RNN;
            v8f acc = {};
            for (int kk = 0; kk < RNN; kk += 32) {
                const v8h  alo = *(const v8h*)(arow + kk + 8 * grp);
                const v8h  ahi = *(const v8h*)(arow + kk + 16 + 8 * grp);
                const v16h a   = __builtin_shufflevector(alo, ahi,
                                     0,1,2,3,4,5,6,7,8,9,10,11,12,13,14,15);
                const v8h  blo = *(const v8h*)(brow + kk + 8 * grp);
                const v8h  bhi = *(const v8h*)(brow + kk + 16 + 8 * grp);
                const v16h bm2 = __builtin_shufflevector(blo, bhi,
                                     0,1,2,3,4,5,6,7,8,9,10,11,12,13,14,15);
                acc = __builtin_amdgcn_wmma_f32_16x16x32_f16(false, a, false, bm2, (short)0,
                                                             acc, false, false);
            }
            const int col = tn * 16 + l15;
#pragma unroll
            for (int v = 0; v < 8; ++v)
                gh_s[(size_t)(8 * grp + v) * G3 + col] = (_Float16)acc[v];
        }
        __syncthreads();

        // fused gate update, one hidden unit per thread-iteration
        for (int u = tid; u < 16 * RNN; u += blockDim.x) {
            const int m = u / RNN, n = u % RNN;
            const size_t grow = ((size_t)(bm + m) * TT + t) * G3;
            const float xr = (float)gx[grow + n];
            const float xz = (float)gx[grow + RNN + n];
            const float xn = (float)gx[grow + 2 * RNN + n];
            const float hr = (float)gh_s[(size_t)m * G3 + n]           + b_hh[n];
            const float hz = (float)gh_s[(size_t)m * G3 + RNN + n]     + b_hh[RNN + n];
            const float hn = (float)gh_s[(size_t)m * G3 + 2 * RNN + n] + b_hh[2 * RNN + n];
            const float r  = sigmoidf_(xr + hr);
            const float z  = sigmoidf_(xz + hz);
            const float nn = tanhf(xn + r * hn);
            const float hprev = (float)h_s[u];
            const float hnew  = (1.f - z) * nn + z * hprev;
            h_s[u] = (_Float16)hnew;
            hs[((size_t)(bm + m) * TT + t) * RNN + n] = (_Float16)hnew;
        }
        __syncthreads();
    }
}

// ---------------------------------------------------------------------------
// Host launcher
// ---------------------------------------------------------------------------
extern "C" void kernel_launch(void* const* d_in, const int* in_sizes, int n_in,
                              void* d_out, int out_size, void* d_ws, size_t ws_size,
                              hipStream_t stream) {
    (void)in_sizes; (void)n_in; (void)out_size; (void)ws_size;
    const float* mels = (const float*)d_in[0];
    const int*   wavs = (const int*)  d_in[1];
    const float* emb  = (const float*)d_in[2];
    const float* wih[4]  = { (const float*)d_in[3],  (const float*)d_in[7],
                             (const float*)d_in[11], (const float*)d_in[15] };
    const float* whhc[4] = { (const float*)d_in[4],  (const float*)d_in[8],
                             (const float*)d_in[12], (const float*)d_in[16] };
    const float* bih[4]  = { (const float*)d_in[5],  (const float*)d_in[9],
                             (const float*)d_in[13], (const float*)d_in[17] };
    const float* bhhc[4] = { (const float*)d_in[6],  (const float*)d_in[10],
                             (const float*)d_in[14], (const float*)d_in[18] };
    const float* w_ih_2 = (const float*)d_in[19];
    const float* w_hh_2 = (const float*)d_in[20];
    const float* b_ih_2 = (const float*)d_in[21];
    const float* b_hh_2 = (const float*)d_in[22];
    const float* fc1_w  = (const float*)d_in[23];
    const float* fc1_b  = (const float*)d_in[24];
    const float* fc2_w  = (const float*)d_in[25];
    const float* fc2_b  = (const float*)d_in[26];

    // workspace carve-up (256B aligned)
    size_t off = 0;
    auto carve = [&](size_t bytes) -> void* {
        void* p = (char*)d_ws + off;
        off = (off + bytes + 255) & ~(size_t)255;
        return p;
    };
    float*     cond_h0 = (float*)    carve((size_t)BB * FPS * 2 * COND * 4);
    float*     cond_h1 = (float*)    carve((size_t)BB * FPS * 2 * COND * 4);
    _Float16*  x16     = (_Float16*) carve((size_t)BB * TT * XDIM * 2);
    _Float16*  wih2h   = (_Float16*) carve((size_t)G3 * XDIM * 2);
    _Float16*  whh2h   = (_Float16*) carve((size_t)G3 * RNN * 2);
    _Float16*  fc1wh   = (_Float16*) carve((size_t)FCD * RNN * 2);
    _Float16*  fc2wh   = (_Float16*) carve((size_t)QD * FCD * 2);
    _Float16*  gx16    = (_Float16*) carve((size_t)BB * TT * G3 * 2);
    _Float16*  hs16    = (_Float16*) carve((size_t)BB * TT * RNN * 2);
    _Float16*  fc1o16  = (_Float16*) carve((size_t)BB * TT * FCD * 2);

    // 1) weight conversion to f16
    {
        int n;
        n = G3 * XDIM; f32_to_f16_kernel<<<(n + 255) / 256, 256, 0, stream>>>(w_ih_2, wih2h, n);
        n = G3 * RNN;  f32_to_f16_kernel<<<(n + 255) / 256, 256, 0, stream>>>(w_hh_2, whh2h, n);
        n = FCD * RNN; f32_to_f16_kernel<<<(n + 255) / 256, 256, 0, stream>>>(fc1_w, fc1wh, n);
        n = QD * FCD;  f32_to_f16_kernel<<<(n + 255) / 256, 256, 0, stream>>>(fc2_w, fc2wh, n);
    }

    // 2) conditioning bi-GRU, 2 layers (stream order gives layer dependency)
    cond_gru_kernel<<<BB, COND, 0, stream>>>(mels, MEL, wih[0], whhc[0], bih[0], bhhc[0],
                                             cond_h0, 0,    0);
    cond_gru_kernel<<<BB, COND, 0, stream>>>(mels, MEL, wih[1], whhc[1], bih[1], bhhc[1],
                                             cond_h0, COND, 1);
    cond_gru_kernel<<<BB, COND, 0, stream>>>(cond_h0, 2 * COND, wih[2], whhc[2], bih[2], bhhc[2],
                                             cond_h1, 0,    0);
    cond_gru_kernel<<<BB, COND, 0, stream>>>(cond_h0, 2 * COND, wih[3], whhc[3], bih[3], bhhc[3],
                                             cond_h1, COND, 1);

    // 3) build main-GRU input x (embedding lookup + NN-upsampled cond), f16
    build_x_kernel<<<BB * TT, EMBD, 0, stream>>>(wavs, emb, cond_h1, x16);

    // 4) gx = x @ w_ih2^T + b_ih2  (51200 x 512 x 2688 WMMA GEMM, f16 out)
    {
        const int tiles = (BB * TT / 16) * (G3 / 16);       // divisible by 8
        wmma_gemm_kernel<_Float16, false><<<tiles / 8, 256, 16 * XDIM * sizeof(_Float16), stream>>>(
            x16, wih2h, b_ih_2, gx16, BB * TT, G3, XDIM);
    }

    // 5) 1600-step recurrence: 2 persistent workgroups, LDS-resident state
    gru_recurrent_kernel<<<2, 1024, (16 * RNN + 16 * G3) * sizeof(_Float16), stream>>>(
        gx16, whh2h, b_hh_2, hs16);

    // 6) FC1 with fused ReLU (51200 x 896 x 512), f16 out
    {
        const int tiles = (BB * TT / 16) * (FCD / 16);
        wmma_gemm_kernel<_Float16, true><<<tiles / 8, 256, 16 * RNN * sizeof(_Float16), stream>>>(
            hs16, fc1wh, fc1_b, fc1o16, BB * TT, FCD, RNN);
    }

    // 7) FC2 (51200 x 512 x 512), f32 out -> d_out
    {
        const int tiles = (BB * TT / 16) * (QD / 16);
        wmma_gemm_kernel<float, false><<<tiles / 8, 256, 16 * FCD * sizeof(_Float16), stream>>>(
            fc1o16, fc2wh, fc2_b, (float*)d_out, BB * TT, QD, FCD);
    }
}
